// VectorQuantizer_6416681140724
// MI455X (gfx1250) — compile-verified, matrix operand order checked by reading, not run on hardware
//
#include <hip/hip_runtime.h>

// ---------------------------------------------------------------------------
// VQ-VAE vector quantizer, MI455X (gfx1250, wave32, WMMA + async LDS copy).
//   d_in[0]: inputs   [N=131072, D=128] f32
//   d_in[1]: codebook [K=1024,   D=128] f32
//   d_out:   [ quantized N*D | vq_loss 1 | indices N ] f32 (indices as float)
// Strategy: f16 WMMA for x.e^T with fused per-row argmin (distance matrix
// never materialized), B tiles staged with GLOBAL_LOAD_ASYNC_TO_LDS_B128
// (ASYNCcnt-tracked, one barrier per chunk), fused gather + deterministic
// loss reduction.
// ---------------------------------------------------------------------------

typedef __attribute__((ext_vector_type(16))) _Float16 v16h;
typedef __attribute__((ext_vector_type(8)))  _Float16 v8h;
typedef __attribute__((ext_vector_type(8)))  float    v8f;
typedef __attribute__((ext_vector_type(4)))  float    v4f;
typedef __attribute__((ext_vector_type(4)))  int      v4i;

#define N_TOT  131072
#define KCODES 1024
#define DDIM   128
#define NCHUNK (KCODES / 16)      // 64 column chunks of 16 codes
#define BSTRIDE 136               // LDS row stride in halfs (128 + 8 pad -> no bank conflicts)

// ---- gfx1250 async global->LDS copy + ASYNCcnt wait -----------------------
#if defined(__AMDGCN__)
__device__ __forceinline__ void async_copy_b128(const void* gsrc, void* ldst) {
#if __has_builtin(__builtin_amdgcn_global_load_async_to_lds_b128)
    // Builtin signature (probe-derived): param0 = AS1 v4i*, param1 = AS3 v4i*.
    typedef __attribute__((address_space(1))) v4i* gv4i_p;
    typedef __attribute__((address_space(3))) v4i* lv4i_p;
    __builtin_amdgcn_global_load_async_to_lds_b128((gv4i_p)gsrc, (lv4i_p)ldst, 0, 0);
#else
    // VDST = LDS byte address (low 32 bits of generic LDS pointer), VADDR = 64b.
    unsigned lds_off = (unsigned)(unsigned long long)ldst;
    asm volatile("global_load_async_to_lds_b128 %0, %1, off"
                 :: "v"(lds_off), "v"(gsrc) : "memory");
#endif
}
__device__ __forceinline__ void wait_asynccnt0() {
#if __has_builtin(__builtin_amdgcn_s_wait_asynccnt)
    __builtin_amdgcn_s_wait_asynccnt(0);
#else
    asm volatile("s_wait_asynccnt 0x0" ::: "memory");
#endif
}
#else
__device__ __forceinline__ void async_copy_b128(const void*, void*) {}
__device__ __forceinline__ void wait_asynccnt0() {}
#endif

// ---- Kernel A: codebook f32 -> f16, plus |e_k|^2 per code -----------------
__global__ __launch_bounds__(128) void vq_prep(const float* __restrict__ cb,
                                               _Float16* __restrict__ cb16,
                                               float* __restrict__ enorms) {
    const int k = blockIdx.x;          // one code row per block
    const int t = threadIdx.x;         // 128 threads = 4 waves
    float v = cb[(size_t)k * DDIM + t];
    cb16[(size_t)k * DDIM + t] = (_Float16)v;
    float s = v * v;
    #pragma unroll
    for (int m = 1; m < 32; m <<= 1) s += __shfl_xor(s, m, 32);
    __shared__ float sw[4];
    if ((t & 31) == 0) sw[t >> 5] = s;
    __syncthreads();
    if (t == 0) enorms[k] = sw[0] + sw[1] + sw[2] + sw[3];
}

// ---- Kernel B: WMMA distances + argmin + gather + loss partials -----------
// Block = 256 threads (8 waves). Each wave owns 32 rows (2 groups of 16).
// Block covers 256 rows; grid = N/256 = 512 blocks.
__global__ __launch_bounds__(256, 1) void vq_main(const float* __restrict__ x,
                                                  const float* __restrict__ cb32,
                                                  const _Float16* __restrict__ cb16,
                                                  const float* __restrict__ enorms,
                                                  float* __restrict__ outq,
                                                  float* __restrict__ outidx,
                                                  float* __restrict__ partials) {
    __shared__ _Float16 lbs[2][16 * BSTRIDE];  // double-buffered 16x128 f16 B tile
    __shared__ int lidx[256];                  // per-row argmin index

    const int tid  = threadIdx.x;
    const int lane = tid & 31;
    const int w    = tid >> 5;
    const int l15  = lane & 15;
    const int hsel = lane >> 4;                // 0: lanes 0-15, 1: lanes 16-31
    const int koff = hsel * 8;                 // K offset per WMMA 16-bit layout
    const int blockRow = blockIdx.x * 256;
    const int waveRow  = blockRow + w * 32;

    // Per-thread staging addresses: thread t moves 16B of each 4KB tile.
    const int srow = tid >> 4;                 // tile row this thread stages
    const int scol = (tid & 15) * 8;           // half offset within row
    _Float16* sdst0 = &lbs[0][srow * BSTRIDE + scol];
    _Float16* sdst1 = &lbs[1][srow * BSTRIDE + scol];
    const _Float16* sg = cb16 + tid * 8;       // matching global source in tile 0

    // ---- Load A fragments: 32 rows x 128 K as f16, ISA 16-bit A layout.
    // v16h elems 0..7 -> K = kc*32 + koff + i ; elems 8..15 -> K = kc*32+16+koff+i
    v16h a[2][4];
    #pragma unroll
    for (int g = 0; g < 2; ++g) {
        const float* rp = x + (size_t)(waveRow + g * 16 + l15) * DDIM;
        #pragma unroll
        for (int kc = 0; kc < 4; ++kc) {
            v8f f0 = *(const v8f*)(rp + kc * 32 + koff);
            v8f f1 = *(const v8f*)(rp + kc * 32 + 16 + koff);
            v16h av;
            #pragma unroll
            for (int i = 0; i < 8; ++i) {
                av[i]     = (_Float16)f0[i];
                av[8 + i] = (_Float16)f1[i];
            }
            a[g][kc] = av;
        }
    }

    float mval[2][8];
    int   midx[2][8];
    #pragma unroll
    for (int g = 0; g < 2; ++g)
        #pragma unroll
        for (int r = 0; r < 8; ++r) { mval[g][r] = 3.0e38f; midx[g][r] = 0; }

    // ---- Stage chunk 0 via async copy, then publish to the block.
    async_copy_b128(sg, sdst0);
    wait_asynccnt0();
    __syncthreads();

    for (int c = 0; c < NCHUNK; ++c) {
        const int cur = c & 1;
        // Kick off next tile into the other buffer: safe, because passing the
        // end-of-(c-1) barrier means no wave still reads buf[cur^1].
        if (c + 1 < NCHUNK)
            async_copy_b128(cb16 + (size_t)(c + 1) * 16 * DDIM + tid * 8,
                            cur ? sdst0 : sdst1);

        v8f acc0 = {}; v8f acc1 = {};
        const _Float16* bbase = &lbs[cur][l15 * BSTRIDE];
        #pragma unroll
        for (int kc = 0; kc < 4; ++kc) {
            v8h b0 = *(const v8h*)(bbase + kc * 32 + koff);
            v8h b1 = *(const v8h*)(bbase + kc * 32 + 16 + koff);
            v16h bv;
            #pragma unroll
            for (int i = 0; i < 8; ++i) { bv[i] = b0[i]; bv[8 + i] = b1[i]; }
            acc0 = __builtin_amdgcn_wmma_f32_16x16x32_f16(false, a[0][kc], false, bv,
                                                          (short)0, acc0, false, false);
            acc1 = __builtin_amdgcn_wmma_f32_16x16x32_f16(false, a[1][kc], false, bv,
                                                          (short)0, acc1, false, false);
        }
        // epilogue: d'(row,col) = |e_col|^2 - 2*dot ; track running argmin
        const int col = c * 16 + l15;
        const float en = enorms[col];
        #pragma unroll
        for (int r = 0; r < 8; ++r) {
            float v0 = en - 2.0f * acc0[r];
            if (v0 < mval[0][r]) { mval[0][r] = v0; midx[0][r] = col; }
            float v1 = en - 2.0f * acc1[r];
            if (v1 < mval[1][r]) { mval[1][r] = v1; midx[1][r] = col; }
        }
        if (c + 1 < NCHUNK) {
            wait_asynccnt0();   // my slice of buf[cur^1] has landed in LDS
            __syncthreads();    // everyone's slice landed; everyone done with buf[cur]
        }
    }

    // ---- Cross-lane argmin within each 16-lane half (first-index tiebreak).
    #pragma unroll
    for (int g = 0; g < 2; ++g) {
        #pragma unroll
        for (int r = 0; r < 8; ++r) {
            float v = mval[g][r]; int ix = midx[g][r];
            #pragma unroll
            for (int m = 1; m < 16; m <<= 1) {
                float ov = __shfl_xor(v, m, 32);
                int   oi = __shfl_xor(ix, m, 32);
                if (ov < v || (ov == v && oi < ix)) { v = ov; ix = oi; }
            }
            if (l15 == r) {  // one lane per half publishes its row
                const int rowLocal = w * 32 + g * 16 + r + hsel * 8;
                lidx[rowLocal] = ix;
                outidx[blockRow + rowLocal] = (float)ix;
            }
        }
    }
    __syncthreads();

    // ---- Fused gather (quantized = codebook[idx]) + loss partial.
    float lsum = 0.0f;
    #pragma unroll 1
    for (int rr = 0; rr < 32; ++rr) {
        const int rowLocal = w * 32 + rr;
        const int ix  = lidx[rowLocal];
        const int row = blockRow + rowLocal;
        v4f q  = *(const v4f*)(cb32 + (size_t)ix  * DDIM + lane * 4);
        v4f xv = *(const v4f*)(x    + (size_t)row * DDIM + lane * 4);
        *(v4f*)(outq + (size_t)row * DDIM + lane * 4) = q;
        #pragma unroll
        for (int i = 0; i < 4; ++i) { float d = q[i] - xv[i]; lsum += d * d; }
    }
    #pragma unroll
    for (int m = 1; m < 32; m <<= 1) lsum += __shfl_xor(lsum, m, 32);
    if (lane == 0) partials[blockIdx.x * 8 + w] = lsum;  // deterministic, no atomics
}

// ---- Kernel C: reduce 4096 per-wave partials -> vq_loss -------------------
__global__ __launch_bounds__(256) void vq_finish(const float* __restrict__ partials,
                                                 float* __restrict__ loss) {
    __shared__ float sh[256];
    const int t = threadIdx.x;
    float s = 0.0f;
    for (int i = t; i < 4096; i += 256) s += partials[i];
    sh[t] = s;
    __syncthreads();
    for (int stride = 128; stride > 0; stride >>= 1) {
        if (t < stride) sh[t] += sh[t + stride];
        __syncthreads();
    }
    // vq_loss = q_loss + 0.25*e_loss = 1.25 * mean((q-x)^2)
    if (t == 0) *loss = sh[0] * (1.25f / ((float)N_TOT * (float)DDIM));
}

extern "C" void kernel_launch(void* const* d_in, const int* in_sizes, int n_in,
                              void* d_out, int out_size, void* d_ws, size_t ws_size,
                              hipStream_t stream) {
    const float* x  = (const float*)d_in[0];   // [N, D]
    const float* cb = (const float*)d_in[1];   // [K, D]

    float* outq   = (float*)d_out;                       // N*D
    float* loss   = outq + (size_t)N_TOT * DDIM;         // 1
    float* outidx = loss + 1;                            // N

    char* ws = (char*)d_ws;
    _Float16* cb16   = (_Float16*)ws;                                  // 256 KB
    float*    enorms = (float*)(ws + (size_t)KCODES * DDIM * 2);       // 4 KB
    float*    parts  = (float*)(ws + (size_t)KCODES * DDIM * 2
                                   + (size_t)KCODES * 4);              // 16 KB

    vq_prep<<<KCODES, 128, 0, stream>>>(cb, cb16, enorms);
    vq_main<<<N_TOT / 256, 256, 0, stream>>>(x, cb, cb16, enorms, outq, outidx, parts);
    vq_finish<<<1, 256, 0, stream>>>(parts, loss);
}
